// AdaptiveMultiSiren_66520453480874
// MI455X (gfx1250) — matrix operand order repaired and checked by reading, not compiled, not run on hardware
//
#include <hip/hip_runtime.h>

// ---------------------------------------------------------------------------
// AdaptiveMultiSiren on MI455X (gfx1250): per-block SIREN MLP via v_wmma
//   x[16x64(pad)] -> (A @ Wt^T tiles) f16 WMMA, f32 accum, sin(30 z) fused.
// ---------------------------------------------------------------------------

typedef __attribute__((ext_vector_type(16))) _Float16 v16h;
typedef __attribute__((ext_vector_type(8)))  _Float16 v8h;
typedef __attribute__((ext_vector_type(8)))  float    v8f;

#define C_      4096
#define T_      1024
#define DH_     64
#define OMEGA0  30.0f
#define WROW    72      // padded LDS row stride in f16 elements (144 B)

union V16 { v16h v; v8h h[2]; };

__device__ __forceinline__ void lds_fence() {
  // compiler barrier + wait for all outstanding LDS ops of this wave
  asm volatile("s_wait_dscnt 0x0" ::: "memory");
}

__device__ __forceinline__ unsigned pack2(float a, float b) {
  union { _Float16 h[2]; unsigned u; } p;
  p.h[0] = (_Float16)a; p.h[1] = (_Float16)b;
  return p.u;
}

__global__ __launch_bounds__(256)
void siren_wmma_kernel(const float* __restrict__ inp,
                       const int*   __restrict__ indices,
                       const float* __restrict__ W0, const float* __restrict__ b0,
                       const float* __restrict__ W1, const float* __restrict__ b1,
                       const float* __restrict__ W2, const float* __restrict__ b2,
                       const float* __restrict__ W3, const float* __restrict__ b3,
                       float* __restrict__ out)
{
  // transposed weights: wt[layer][o][i] as f16 (layer0 zero-padded for i>=3)
  __shared__ __align__(16) _Float16 wt[3][64 * WROW];
  __shared__ __align__(16) _Float16 w3t[16 * WROW];        // [n(pad16)][i]
  __shared__ __align__(16) _Float16 act[8][16 * WROW];     // per-wave 16x64 tile
  __shared__ float biasL[3][64];
  __shared__ float bias3[16];

  const int tid   = threadIdx.x;
  const int lane  = tid & 31;
  const int wave  = tid >> 5;
  const int l15   = lane & 15;
  const int lhalf = lane >> 4;              // 0: lanes 0-15, 1: lanes 16-31
  const int hoffA = lhalf ? 8 : 0;          // A-frag half-wave K offset
  const int blk   = blockIdx.x;
  const int idx   = indices[blk];

  // ---------------- stage weights (gather + f32->f16 + transpose) ----------
  // zero pad regions first
  for (int e = tid; e < 64 * 32; e += 256) {           // layer0: i in [0,32)
    int o = e >> 5, i = e & 31;
    wt[0][o * WROW + i] = (_Float16)0.f;
  }
  for (int e = tid; e < 16 * 64; e += 256) {           // W3t full pad
    int n = e >> 6, i = e & 63;
    w3t[n * WROW + i] = (_Float16)0.f;
  }
  if (tid < 16) bias3[tid] = (tid < 3) ? b3[(size_t)idx * 3 + tid] : 0.f;
  __syncthreads();

  for (int e = tid; e < 3 * 64; e += 256) {            // W0: [3][64]
    int i = e >> 6, o = e & 63;
    wt[0][o * WROW + i] = (_Float16)W0[(size_t)idx * 3 * 64 + e];
  }
  for (int e = tid; e < 64 * 64; e += 256) {           // W1: [64][64]
    int i = e >> 6, o = e & 63;
    wt[1][o * WROW + i] = (_Float16)W1[(size_t)idx * 64 * 64 + e];
  }
  for (int e = tid; e < 64 * 64; e += 256) {           // W2: [64][64]
    int i = e >> 6, o = e & 63;
    wt[2][o * WROW + i] = (_Float16)W2[(size_t)idx * 64 * 64 + e];
  }
  for (int e = tid; e < 64 * 3; e += 256) {            // W3: [64][3]
    int i = e / 3, n = e - i * 3;
    w3t[n * WROW + i] = (_Float16)W3[(size_t)idx * 64 * 3 + e];
  }
  for (int e = tid; e < 3 * 64; e += 256) {            // hidden biases
    int l = e >> 6, o = e & 63;
    const float* bp = (l == 0) ? b0 : ((l == 1) ? b1 : b2);
    biasL[l][o] = bp[(size_t)idx * 64 + o];
  }
  __syncthreads();

  _Float16* abuf = act[wave];

  // ---------------- main loop: 8 row-tiles of 16 coords per wave -----------
  for (int it = 0; it < 8; ++it) {
    const int t0 = (wave + it * 8) * 16;

    // fill activation tile with x (cols 0..2), zero cols 3..31 (layer0 K=32)
    {
      const int r = l15;
      unsigned* rowp = (unsigned*)&abuf[r * WROW];
      if (lhalf == 0) {
        const float* xp = inp + ((size_t)idx * T_ + (size_t)(t0 + r)) * 3;
        rowp[0] = pack2(xp[0], xp[1]);
        rowp[1] = pack2(xp[2], 0.f);
#pragma unroll
        for (int d = 2; d < 8; ++d) rowp[d] = 0u;
      } else {
#pragma unroll
        for (int d = 8; d < 16; ++d) rowp[d] = 0u;
      }
    }
    lds_fence();

    // ---- three sine layers: h = sin(30 * (A @ W + b)) ----
#pragma unroll
    for (int layer = 0; layer < 3; ++layer) {
      const int nk = (layer == 0) ? 1 : 2;

      // hoist A fragments (16x32 f16 layout: lanes0-15 M rows, K pairs)
      V16 afr[2];
      for (int kk = 0; kk < nk; ++kk) {
        const _Float16* ap = abuf + l15 * WROW + kk * 32 + hoffA;
        afr[kk].h[0] = *(const v8h*)(ap);        // K = kk*32 + {0..7|8..15}
        afr[kk].h[1] = *(const v8h*)(ap + 16);   // K = kk*32 + 16 + {...}
      }

      const _Float16* wl = wt[layer];
      v8f accs[4];
#pragma unroll
      for (int nt = 0; nt < 4; ++nt) {
        v8f c = {};
        for (int kk = 0; kk < nk; ++kk) {
          const int n = nt * 16 + l15;
          // B frag: lanes0-15 K=kk*32+{0..7,16..23}, lanes16-31 K=kk*32+{8..15,24..31}, N=l15
          const _Float16* bp = wl + n * WROW + kk * 32 + lhalf * 8;
          V16 bfr;
          bfr.h[0] = *(const v8h*)(bp);
          bfr.h[1] = *(const v8h*)(bp + 16);
          c = __builtin_amdgcn_wmma_f32_16x16x32_f16(
              false, afr[kk].v, false, bfr.v, (short)0, c, false, false);
        }
        accs[nt] = c;
      }

      // bias + sin(30 z), write f16 back into the tile (D: N=l15, M=lhalf*8+v)
#pragma unroll
      for (int nt = 0; nt < 4; ++nt) {
        const int n = nt * 16 + l15;
        const float bia = biasL[layer][n];
#pragma unroll
        for (int v = 0; v < 8; ++v) {
          const int mrow = lhalf * 8 + v;
          float z = OMEGA0 * (accs[nt][v] + bia);
          abuf[mrow * WROW + n] = (_Float16)__sinf(z);
        }
      }
      lds_fence();
    }

    // ---- output layer: [16x64] @ W3[64x3] + b3 (N padded to 16) ----
    {
      V16 afr[2];
#pragma unroll
      for (int kk = 0; kk < 2; ++kk) {
        const _Float16* ap = abuf + l15 * WROW + kk * 32 + hoffA;
        afr[kk].h[0] = *(const v8h*)(ap);
        afr[kk].h[1] = *(const v8h*)(ap + 16);
      }
      v8f c = {};
#pragma unroll
      for (int kk = 0; kk < 2; ++kk) {
        const _Float16* bp = w3t + l15 * WROW + kk * 32 + lhalf * 8;
        V16 bfr;
        bfr.h[0] = *(const v8h*)(bp);
        bfr.h[1] = *(const v8h*)(bp + 16);
        c = __builtin_amdgcn_wmma_f32_16x16x32_f16(
            false, afr[kk].v, false, bfr.v, (short)0, c, false, false);
      }
      const int n = l15;
      if (n < 3) {
        const float bia = bias3[n];
#pragma unroll
        for (int v = 0; v < 8; ++v) {
          const int mrow = lhalf * 8 + v;
          out[((size_t)blk * T_ + (size_t)(t0 + mrow)) * 3 + n] = c[v] + bia;
        }
      }
    }
    lds_fence();   // barrier before next iteration's x-fill overwrites tile
  }
}

// ---------------------------------------------------------------------------
extern "C" void kernel_launch(void* const* d_in, const int* in_sizes, int n_in,
                              void* d_out, int out_size, void* d_ws, size_t ws_size,
                              hipStream_t stream) {
  const float* inp     = (const float*)d_in[0];
  const int*   indices = (const int*)  d_in[1];
  const float* W0 = (const float*)d_in[2];
  const float* b0 = (const float*)d_in[3];
  const float* W1 = (const float*)d_in[4];
  const float* b1 = (const float*)d_in[5];
  const float* W2 = (const float*)d_in[6];
  const float* b2 = (const float*)d_in[7];
  const float* W3 = (const float*)d_in[8];
  const float* b3 = (const float*)d_in[9];
  float* out = (float*)d_out;

  const int B = in_sizes[1];   // number of active blocks (2048)
  dim3 grid(B), block(256);
  hipLaunchKernelGGL(siren_wmma_kernel, grid, block, 0, stream,
                     inp, indices, W0, b0, W1, b1, W2, b2, W3, b3, out);
}
